// Encoder_55362128445863
// MI455X (gfx1250) — matrix operand order repaired and checked by reading, not compiled, not run on hardware
//
#include <hip/hip_runtime.h>
#include <hip/hip_bf16.h>

#define H_ 75
#define E_ 75
#define V_ 75
#define P_LEN_ 768
#define Q_LEN_ 384

typedef __attribute__((ext_vector_type(16))) _Float16 v16h;
typedef __attribute__((ext_vector_type(8)))  float    v8f;

// Branch-free fast sigmoid / tanh: native v_exp_f32 + v_rcp_f32, no EXEC games.
__device__ __forceinline__ float sigf(float x) {
  return __builtin_amdgcn_rcpf(1.0f + __expf(-x));
}
__device__ __forceinline__ float tanh_fast(float x) {
  return 2.0f * __builtin_amdgcn_rcpf(1.0f + __expf(-2.0f * x)) - 1.0f;
}

// --------------------------------------------------------------------------
// Embedding gather: p_emb[768,75], q_emb[384,75]
// --------------------------------------------------------------------------
__global__ void k_embed(const int* __restrict__ pas, const int* __restrict__ que,
                        const float* __restrict__ embed,
                        float* __restrict__ p_emb, float* __restrict__ q_emb) {
  int i = blockIdx.x * blockDim.x + threadIdx.x;
  const int totP = P_LEN_ * E_;
  const int totQ = Q_LEN_ * E_;
  if (i < totP) {
    int t = i / E_, e = i % E_;
    p_emb[i] = embed[pas[t] * E_ + e];
  } else {
    int j = i - totP;
    if (j < totQ) {
      int t = j / E_, e = j % E_;
      q_emb[j] = embed[que[t] * E_ + e];
    }
  }
}

// --------------------------------------------------------------------------
// Generic WMMA GEMM: C[M,N] = act( A[M,K] @ W[N,K]^T + bias[N] )
// f32 operands in global, f16 fragments in-register, f32 WMMA accumulate.
// PRE_TANH applies tanh to A elements. ACT: 0 = none, 1 = sigmoid.
// --------------------------------------------------------------------------
template <int PRE_TANH, int ACT>
__global__ void __launch_bounds__(256) k_gemm(const float* __restrict__ A,
                                              const float* __restrict__ W,
                                              const float* __restrict__ bias,
                                              float* __restrict__ C,
                                              int M, int N, int K) {
  const int lane   = threadIdx.x & 31;
  const int wave   = threadIdx.x >> 5;
  const int nwaves = blockDim.x >> 5;
  const int mtiles = (M + 15) / 16;
  const int ntiles = (N + 15) / 16;
  const int tiles  = mtiles * ntiles;
  const int kcn    = (K + 31) / 32;

  for (int tile = blockIdx.x * nwaves + wave; tile < tiles; tile += gridDim.x * nwaves) {
    const int mt = tile / ntiles;
    const int nt = tile % ntiles;
    v8f acc = {};
    const int m = mt * 16 + (lane & 15);
    const int n = nt * 16 + (lane & 15);
    for (int kc = 0; kc < kcn; ++kc) {
      v16h af, bf;
      // A fragment: lanes 0-15 -> K 0..7 / 16..23 ; lanes 16-31 -> K 8..15 / 24..31
      const int kbA = kc * 32 + ((lane < 16) ? 0 : 8);
#pragma unroll
      for (int j = 0; j < 16; ++j) {
        const int k = kbA + ((j < 8) ? j : (8 + j));
        float v = 0.0f;
        if (m < M && k < K) v = A[m * K + k];
        if (PRE_TANH) v = tanh_fast(v);
        af[j] = (_Float16)v;
      }
      // B fragment: lane = column n; lanes 0-15 K 0..7 / 16..23 ; lanes 16-31 K 8..15 / 24..31
      const int kbB = kc * 32 + ((lane < 16) ? 0 : 8);
#pragma unroll
      for (int j = 0; j < 16; ++j) {
        const int k = kbB + ((j < 8) ? j : (8 + j));
        float v = 0.0f;
        if (n < N && k < K) v = W[n * K + k];
        bf[j] = (_Float16)v;
      }
      acc = __builtin_amdgcn_wmma_f32_16x16x32_f16(false, af, false, bf, (short)0, acc,
                                                   false, false);
    }
    const int m0 = mt * 16 + ((lane < 16) ? 0 : 8);
#pragma unroll
    for (int i = 0; i < 8; ++i) {
      const int mm = m0 + i;
      if (mm < M && n < N) {
        float v = acc[i] + bias[n];
        if (ACT == 1) v = sigf(v);
        C[mm * N + n] = v;
      }
    }
  }
}

// --------------------------------------------------------------------------
// Encoder GRU time scans: 4 blocks = {passage-fwd, passage-bwd, q-fwd, q-bwd}
// --------------------------------------------------------------------------
__global__ void __launch_bounds__(256) k_gru_scan(
    const float* __restrict__ giPF, const float* __restrict__ giPB,
    const float* __restrict__ giQF, const float* __restrict__ giQB,
    const float* __restrict__ WhhF, const float* __restrict__ bhhF,
    const float* __restrict__ WhhB, const float* __restrict__ bhhB,
    float* __restrict__ outP, float* __restrict__ outQ) {
  __shared__ float h[80], gh[232], hnew[80];
  const int tid = threadIdx.x;
  const int b = blockIdx.x;
  const float* gi  = (b == 0) ? giPF : (b == 1) ? giPB : (b == 2) ? giQF : giQB;
  const float* Whh = (b & 1) ? WhhB : WhhF;
  const float* bhh = (b & 1) ? bhhB : bhhF;
  float* out = (b < 2) ? outP : outQ;
  const int T = (b < 2) ? P_LEN_ : Q_LEN_;
  const int rev = b & 1;
  const int colOff = (b & 1) ? H_ : 0;

  if (tid < 80) h[tid] = 0.0f;
  __syncthreads();
  for (int sstep = 0; sstep < T; ++sstep) {
    const int t = rev ? (T - 1 - sstep) : sstep;
    if (tid < 225) {
      float a = bhh[tid];
      const float* w = Whh + tid * H_;
      for (int k = 0; k < H_; ++k) a += w[k] * h[k];
      gh[tid] = a;
    }
    __syncthreads();
    if (tid < H_) {
      const float* g = gi + t * 225;
      float r = sigf(g[tid] + gh[tid]);
      float z = sigf(g[75 + tid] + gh[75 + tid]);
      float n = tanh_fast(g[150 + tid] + r * gh[150 + tid]);
      float hv = (1.0f - z) * n + z * h[tid];
      hnew[tid] = hv;
      out[t * 150 + colOff + tid] = hv;
    }
    __syncthreads();
    if (tid < H_) h[tid] = hnew[tid];
    __syncthreads();
  }
}

// --------------------------------------------------------------------------
// Pack Wvt (75x75) into WMMA B-fragment layout in LDS, once per kernel:
// bpack[(nt*3+kc)*32 + lane] = 16 halves for that fragment/lane.
// --------------------------------------------------------------------------
__device__ void pack_b(const float* __restrict__ Wvt, v16h* bpack) {
  _Float16* bh = (_Float16*)bpack;
  for (int idx = threadIdx.x; idx < 480 * 16; idx += blockDim.x) {
    const int frag = idx >> 4;
    const int j    = idx & 15;
    const int lane = frag & 31;
    const int ntkc = frag >> 5;  // 0..14
    const int nt = ntkc / 3, kc = ntkc % 3;
    const int n = nt * 16 + (lane & 15);
    const int k = kc * 32 + ((lane < 16) ? 0 : 8) + ((j < 8) ? j : (8 + j));
    float v = (n < V_ && k < V_) ? Wvt[n * V_ + k] : 0.0f;
    bh[idx] = (_Float16)v;
  }
}

// --------------------------------------------------------------------------
// Block-wide WMMA attention + fused reduction:
//   a[m,n] = sigmoid( (tanh(Qmat + s[None,:]) @ Wvt^T)[m,n] + bvt[n] )
//   c_t[n] += sum_m a[m,n] * Qmat[m,n]
// One row-tile per wave iteration: build 3 tanh'd A-fragments once, fire
// 15 WMMAs against LDS-prepacked B fragments (5x A reuse).
// --------------------------------------------------------------------------
template <int MROWS>
__device__ void attention_ct(const float* __restrict__ Qmat, const float* __restrict__ s,
                             const v16h* bpack, const float* __restrict__ bvt,
                             float* c_t /* LDS, zeroed */) {
  const int lane   = threadIdx.x & 31;
  const int wave   = threadIdx.x >> 5;
  const int nwaves = blockDim.x >> 5;
  const int mtiles = MROWS / 16;
  for (int mt = wave; mt < mtiles; mt += nwaves) {
    const int m = mt * 16 + (lane & 15);
    v16h af[3];
#pragma unroll
    for (int kc = 0; kc < 3; ++kc) {
      const int kbA = kc * 32 + ((lane < 16) ? 0 : 8);
#pragma unroll
      for (int j = 0; j < 16; ++j) {
        const int k = kbA + ((j < 8) ? j : (8 + j));
        float v = 0.0f;
        if (k < V_) v = tanh_fast(Qmat[m * V_ + k] + s[k]);
        af[kc][j] = (_Float16)v;
      }
    }
    const int nn = lane & 15;
    const int m0 = mt * 16 + ((lane < 16) ? 0 : 8);
#pragma unroll
    for (int nt = 0; nt < 5; ++nt) {
      v8f acc = {};
#pragma unroll
      for (int kc = 0; kc < 3; ++kc) {
        acc = __builtin_amdgcn_wmma_f32_16x16x32_f16(
            false, af[kc], false, bpack[(nt * 3 + kc) * 32 + lane], (short)0, acc,
            false, false);
      }
      const int n = nt * 16 + nn;
      if (n < V_) {
        const float bn = bvt[n];
        float part = 0.0f;
#pragma unroll
        for (int i = 0; i < 8; ++i) {
          part += sigf(acc[i] + bn) * Qmat[(m0 + i) * V_ + n];
        }
        atomicAdd(&c_t[n], part);
      }
    }
  }
}

// --------------------------------------------------------------------------
// One stacked-GRU layer step (block-wide). inp/h/gi/gh/hnew are LDS.
// --------------------------------------------------------------------------
__device__ void gru_layer(const float* __restrict__ Wih, const float* __restrict__ bih,
                          const float* __restrict__ Whh, const float* __restrict__ bhh,
                          const float* inp, int inDim, float* h,
                          float* gi, float* gh, float* hnew) {
  const int tid = threadIdx.x;
  if (tid < 225) {
    float a = bih[tid];
    const float* w = Wih + tid * inDim;
    for (int k = 0; k < inDim; ++k) a += w[k] * inp[k];
    gi[tid] = a;
    float b = bhh[tid];
    const float* w2 = Whh + tid * H_;
    for (int k = 0; k < H_; ++k) b += w2[k] * h[k];
    gh[tid] = b;
  }
  __syncthreads();
  if (tid < H_) {
    float r = sigf(gi[tid] + gh[tid]);
    float z = sigf(gi[H_ + tid] + gh[H_ + tid]);
    float n = tanh_fast(gi[2 * H_ + tid] + r * gh[2 * H_ + tid]);
    hnew[tid] = (1.0f - z) * n + z * h[tid];
  }
  __syncthreads();
  if (tid < H_) h[tid] = hnew[tid];
  __syncthreads();
}

// --------------------------------------------------------------------------
// Question-aware passage scan: 768 sequential steps, persistent 1 workgroup.
// --------------------------------------------------------------------------
__global__ void __launch_bounds__(256) k_qa_scan(
    const float* __restrict__ q_l, const float* __restrict__ u_p,
    const float* __restrict__ plin,
    const float* Wh, const float* bh, const float* Wvt, const float* bvt,
    const float* Wga, const float* bga,
    const float* Wih0, const float* Whh0, const float* bih0, const float* bhh0,
    const float* Wih1, const float* Whh1, const float* bih1, const float* bhh1,
    const float* Wih2, const float* Whh2, const float* bih2, const float* bhh2,
    float* __restrict__ vout) {
  __shared__ v16h bpack[480];
  __shared__ float s[80], c_t[80], fi[232], gated[232];
  __shared__ float h0[80], h1[80], h2[80], gi[232], gh[232], hnew[80];
  const int tid = threadIdx.x;
  pack_b(Wvt, bpack);
  if (tid < 80) { h0[tid] = 0.0f; h1[tid] = 0.0f; h2[tid] = 0.0f; }
  __syncthreads();
  for (int t = 0; t < P_LEN_; ++t) {
    if (tid < V_) {
      float a = bh[tid];
      const float* w = Wh + tid * H_;
      for (int k = 0; k < H_; ++k) a += w[k] * h2[k];
      s[tid] = a + plin[t * V_ + tid];
    }
    if (tid < 80) c_t[tid] = 0.0f;
    __syncthreads();
    attention_ct<Q_LEN_>(q_l, s, bpack, bvt, c_t);
    __syncthreads();
    if (tid < 150) fi[tid] = u_p[t * 150 + tid];
    else if (tid < 225) fi[tid] = c_t[tid - 150];
    __syncthreads();
    if (tid < 225) {
      float a = bga[tid];
      const float* w = Wga + tid * 225;
      for (int k = 0; k < 225; ++k) a += w[k] * fi[k];
      gated[tid] = fi[tid] * sigf(a);
    }
    __syncthreads();
    gru_layer(Wih0, bih0, Whh0, bhh0, gated, 225, h0, gi, gh, hnew);
    gru_layer(Wih1, bih1, Whh1, bhh1, h0, H_, h1, gi, gh, hnew);
    gru_layer(Wih2, bih2, Whh2, bhh2, h1, H_, h2, gi, gh, hnew);
    if (tid < V_) vout[t * V_ + tid] = h2[tid];
    __syncthreads();
  }
}

// --------------------------------------------------------------------------
// Self-matching attention scan: 768 sequential steps, persistent 1 workgroup.
// --------------------------------------------------------------------------
__global__ void __launch_bounds__(256) k_sm_scan(
    const float* __restrict__ p_full, const float* __restrict__ vin,
    const float* Wsw, const float* bsw, const float* Wvt, const float* bvt,
    const float* Wga, const float* bga,
    const float* Wih0, const float* Whh0, const float* bih0, const float* bhh0,
    const float* Wih1, const float* Whh1, const float* bih1, const float* bhh1,
    const float* Wih2, const float* Whh2, const float* bih2, const float* bhh2,
    float* __restrict__ hmat) {
  __shared__ v16h bpack[480];
  __shared__ float pw[80], s[80], c_t[80], fi[232], gated[232];
  __shared__ float h0[80], h1[80], h2[80], gi[232], gh[232], hnew[80];
  const int tid = threadIdx.x;
  pack_b(Wvt, bpack);
  if (tid < 80) { h0[tid] = 0.0f; h1[tid] = 0.0f; h2[tid] = 0.0f; }
  __syncthreads();
  for (int t = 0; t < P_LEN_; ++t) {
    if (tid < V_) pw[tid] = vin[t * V_ + tid];
    __syncthreads();
    if (tid < V_) {
      float a = bsw[tid];
      const float* w = Wsw + tid * V_;
      for (int k = 0; k < V_; ++k) a += w[k] * pw[k];
      s[tid] = a;
    }
    if (tid < 80) c_t[tid] = 0.0f;
    __syncthreads();
    attention_ct<P_LEN_>(p_full, s, bpack, bvt, c_t);
    __syncthreads();
    if (tid < V_) fi[tid] = pw[tid];
    else if (tid < 150) fi[tid] = c_t[tid - V_];
    __syncthreads();
    if (tid < 150) {
      float a = bga[tid];
      const float* w = Wga + tid * 150;
      for (int k = 0; k < 150; ++k) a += w[k] * fi[k];
      gated[tid] = fi[tid] * sigf(a);
    }
    __syncthreads();
    gru_layer(Wih0, bih0, Whh0, bhh0, gated, 150, h0, gi, gh, hnew);
    gru_layer(Wih1, bih1, Whh1, bhh1, h0, H_, h1, gi, gh, hnew);
    gru_layer(Wih2, bih2, Whh2, bhh2, h1, H_, h2, gi, gh, hnew);
    if (tid < V_) hmat[t * V_ + tid] = h2[tid];
    __syncthreads();
  }
}

// --------------------------------------------------------------------------
// Pointer network: h_ans init reduction + two pointer steps (argmax + GRUCell)
// --------------------------------------------------------------------------
__global__ void __launch_bounds__(256) k_pointer(
    const float* __restrict__ a0, const float* __restrict__ q_l,
    const float* __restrict__ hp,
    const float* Wha, const float* bha, const float* wvta, const float* bvta,
    const float* Wih, const float* Whh, const float* bih, const float* bhh,
    float* __restrict__ out) {
  __shared__ float h_ans[80], h_a[80], c_t[80], gi[232], gh[232], hnew[80];
  __shared__ float ascore[P_LEN_];
  const int tid = threadIdx.x;
  if (tid < V_) {
    float acc = 0.0f;
    for (int m = 0; m < Q_LEN_; ++m) acc += a0[m * V_ + tid] * q_l[m * V_ + tid];
    h_ans[tid] = acc;
  }
  __syncthreads();
  for (int it = 0; it < 2; ++it) {
    if (tid < V_) {
      float a = bha[tid];
      const float* w = Wha + tid * V_;
      for (int k = 0; k < V_; ++k) a += w[k] * h_ans[k];
      h_a[tid] = a;
    }
    __syncthreads();
    for (int p = tid; p < P_LEN_; p += 256) {
      float sc = bvta[0];
      const float* hrow = hp + p * V_;
      for (int k = 0; k < V_; ++k) sc += wvta[k] * tanh_fast(hrow[k] + h_a[k]);
      ascore[p] = sigf(sc);
    }
    __syncthreads();
    if (tid == 0) {
      float best = ascore[0]; int bi = 0;
      for (int p = 1; p < P_LEN_; ++p)
        if (ascore[p] > best) { best = ascore[p]; bi = p; }
      out[it] = (float)bi;
    }
    if (tid < V_) {
      float acc = 0.0f;
      for (int p = 0; p < P_LEN_; ++p) acc += ascore[p] * hp[p * V_ + tid];
      c_t[tid] = acc;
    }
    __syncthreads();
    if (tid < 225) {
      float a = bih[tid];
      const float* w = Wih + tid * V_;
      for (int k = 0; k < V_; ++k) a += w[k] * c_t[k];
      gi[tid] = a;
      float b = bhh[tid];
      const float* w2 = Whh + tid * H_;
      for (int k = 0; k < H_; ++k) b += w2[k] * h_ans[k];
      gh[tid] = b;
    }
    __syncthreads();
    if (tid < V_) {
      float r = sigf(gi[tid] + gh[tid]);
      float z = sigf(gi[75 + tid] + gh[75 + tid]);
      float n = tanh_fast(gi[150 + tid] + r * gh[150 + tid]);
      hnew[tid] = (1.0f - z) * n + z * h_ans[tid];
    }
    __syncthreads();
    if (tid < V_) h_ans[tid] = hnew[tid];
    __syncthreads();
  }
}

// --------------------------------------------------------------------------
#define GEMM_GRID(M, N) (((((M) + 15) / 16) * (((N) + 15) / 16) + 7) / 8)

extern "C" void kernel_launch(void* const* d_in, const int* in_sizes, int n_in,
                              void* d_out, int out_size, void* d_ws, size_t ws_size,
                              hipStream_t stream) {
  (void)in_sizes; (void)n_in; (void)out_size; (void)ws_size;

  // ---- input leaf mapping (setup_inputs() insertion order) ----
  const int*   passage  = (const int*)d_in[0];
  const int*   question = (const int*)d_in[1];
  const float* embed    = (const float*)d_in[2];
  auto ENC = [&](int l, int d, int k) { return (const float*)d_in[3 + (l * 2 + d) * 4 + k]; };
  const float* W_passage  = (const float*)d_in[27]; const float* b_passage  = (const float*)d_in[28];
  const float* W_question = (const float*)d_in[29]; const float* b_question = (const float*)d_in[30];
  const float* W_hidden   = (const float*)d_in[31]; const float* b_hidden   = (const float*)d_in[32];
  const float* W_vt       = (const float*)d_in[33]; const float* b_vt       = (const float*)d_in[34];
  const float* W_vt_ans   = (const float*)d_in[35]; const float* b_vt_ans   = (const float*)d_in[36];
  const float* W_sp_word  = (const float*)d_in[37]; const float* b_sp_word  = (const float*)d_in[38];
  const float* W_sp       = (const float*)d_in[39]; const float* b_sp       = (const float*)d_in[40];
  const float* W_h_p      = (const float*)d_in[41]; const float* b_h_p      = (const float*)d_in[42];
  const float* W_h_a      = (const float*)d_in[43]; const float* b_h_a      = (const float*)d_in[44];
  const float* W_ga       = (const float*)d_in[45]; const float* b_ga       = (const float*)d_in[46];
  const float* W_ga_sm    = (const float*)d_in[47]; const float* b_ga_sm    = (const float*)d_in[48];
  auto QA = [&](int l, int k) { return (const float*)d_in[49 + l * 4 + k]; };
  auto SM = [&](int l, int k) { return (const float*)d_in[61 + l * 4 + k]; };
  const float* ans_Wih = (const float*)d_in[73];
  const float* ans_Whh = (const float*)d_in[74];
  const float* ans_bih = (const float*)d_in[75];
  const float* ans_bhh = (const float*)d_in[76];

  // ---- workspace layout (floats) ----
  float* ws = (float*)d_ws;
  float* p_emb  = ws;               // 768*75
  float* q_emb  = p_emb + 57600;    // 384*75
  float* bufPA  = q_emb + 28800;    // 768*150
  float* bufPB  = bufPA + 115200;
  float* bufQA  = bufPB + 115200;   // 384*150
  float* bufQB  = bufQA + 57600;
  float* giPF   = bufQB + 57600;    // 768*225
  float* giPB   = giPF + 172800;
  float* giQF   = giPB + 172800;    // 384*225
  float* giQB   = giQF + 86400;
  float* q_l    = giQB + 86400;     // 384*75
  float* plin   = q_l + 28800;      // 768*75
  float* a0     = plin + 57600;     // 384*75
  float* vbuf   = a0 + 28800;       // 768*75
  float* p_full = vbuf + 57600;     // 768*75
  float* hmat   = p_full + 57600;   // 768*75
  float* hp_lin = hmat + 57600;     // 768*75

  // ---- Phase A: embeddings ----
  {
    int tot = (P_LEN_ + Q_LEN_) * E_;
    k_embed<<<(tot + 255) / 256, 256, 0, stream>>>(passage, question, embed, p_emb, q_emb);
  }

  // ---- Phase B: 3-layer BiGRU encoder (GI via WMMA GEMM, 4 concurrent scans) ----
  const float* inP = p_emb;
  const float* inQ = q_emb;
  float* outP = bufPA;
  float* outQ = bufQA;
  for (int l = 0; l < 3; ++l) {
    int inDim = (l == 0) ? E_ : 150;
    k_gemm<0, 0><<<GEMM_GRID(P_LEN_, 225), 256, 0, stream>>>(inP, ENC(l, 0, 0), ENC(l, 0, 2), giPF, P_LEN_, 225, inDim);
    k_gemm<0, 0><<<GEMM_GRID(P_LEN_, 225), 256, 0, stream>>>(inP, ENC(l, 1, 0), ENC(l, 1, 2), giPB, P_LEN_, 225, inDim);
    k_gemm<0, 0><<<GEMM_GRID(Q_LEN_, 225), 256, 0, stream>>>(inQ, ENC(l, 0, 0), ENC(l, 0, 2), giQF, Q_LEN_, 225, inDim);
    k_gemm<0, 0><<<GEMM_GRID(Q_LEN_, 225), 256, 0, stream>>>(inQ, ENC(l, 1, 0), ENC(l, 1, 2), giQB, Q_LEN_, 225, inDim);
    k_gru_scan<<<4, 256, 0, stream>>>(giPF, giPB, giQF, giQB,
                                      ENC(l, 0, 1), ENC(l, 0, 3), ENC(l, 1, 1), ENC(l, 1, 3),
                                      outP, outQ);
    inP = outP; inQ = outQ;
    outP = (outP == bufPA) ? bufPB : bufPA;
    outQ = (outQ == bufQA) ? bufQB : bufQA;
  }
  const float* u_p = inP;
  const float* u_q = inQ;

  // ---- Phase C: q_l, plin, a0 (WMMA GEMMs) ----
  k_gemm<0, 0><<<GEMM_GRID(Q_LEN_, V_), 256, 0, stream>>>(u_q, W_question, b_question, q_l, Q_LEN_, V_, 150);
  k_gemm<0, 0><<<GEMM_GRID(P_LEN_, V_), 256, 0, stream>>>(u_p, W_passage, b_passage, plin, P_LEN_, V_, 150);
  k_gemm<1, 1><<<GEMM_GRID(Q_LEN_, V_), 256, 0, stream>>>(q_l, W_vt, b_vt, a0, Q_LEN_, V_, V_);

  // ---- Phase D: question-aware passage scan (persistent, WMMA attention) ----
  k_qa_scan<<<1, 256, 0, stream>>>(q_l, u_p, plin,
                                   W_hidden, b_hidden, W_vt, b_vt, W_ga, b_ga,
                                   QA(0, 0), QA(0, 1), QA(0, 2), QA(0, 3),
                                   QA(1, 0), QA(1, 1), QA(1, 2), QA(1, 3),
                                   QA(2, 0), QA(2, 1), QA(2, 2), QA(2, 3),
                                   vbuf);

  // ---- Phase E: self-matching scan ----
  k_gemm<0, 0><<<GEMM_GRID(P_LEN_, V_), 256, 0, stream>>>(vbuf, W_sp, b_sp, p_full, P_LEN_, V_, V_);
  k_sm_scan<<<1, 256, 0, stream>>>(p_full, vbuf,
                                   W_sp_word, b_sp_word, W_vt, b_vt, W_ga_sm, b_ga_sm,
                                   SM(0, 0), SM(0, 1), SM(0, 2), SM(0, 3),
                                   SM(1, 0), SM(1, 1), SM(1, 2), SM(1, 3),
                                   SM(2, 0), SM(2, 1), SM(2, 2), SM(2, 3),
                                   hmat);

  // ---- Phase F: pointer network ----
  k_gemm<0, 0><<<GEMM_GRID(P_LEN_, V_), 256, 0, stream>>>(hmat, W_h_p, b_h_p, hp_lin, P_LEN_, V_, V_);
  k_pointer<<<1, 256, 0, stream>>>(a0, q_l, hp_lin,
                                   W_h_a, b_h_a, W_vt_ans, b_vt_ans,
                                   ans_Wih, ans_Whh, ans_bih, ans_bhh,
                                   (float*)d_out);
}
